// GraphAttentionLayer_81905026334735
// MI455X (gfx1250) — matrix operand order repaired and checked by reading, not compiled, not run on hardware
//
#include <hip/hip_runtime.h>
#include <hip/hip_bf16.h>

#define N_NODES 8192
#define IN_F    512
#define OUT_F   512
#define ALPHA   0.2f
#define PSTR    40            // LDS P-tile row stride in halves (80B: 16B aligned chunks)

typedef __attribute__((ext_vector_type(16))) __bf16 v16bf;
typedef __attribute__((ext_vector_type(8)))  __bf16 v8bf;
typedef __attribute__((ext_vector_type(8)))  float  v8f;

static __device__ __forceinline__ __bf16 f2bf(float x) {
  unsigned u = __builtin_bit_cast(unsigned, x);
  unsigned r = u + 0x7fffu + ((u >> 16) & 1u);      // round to nearest even
  unsigned short h = (unsigned short)(r >> 16);
  return __builtin_bit_cast(__bf16, h);
}
static __device__ __forceinline__ float bf2f(__bf16 x) {
  unsigned short h = __builtin_bit_cast(unsigned short, x);
  unsigned u = ((unsigned)h) << 16;
  return __builtin_bit_cast(float, u);
}

// ---------------- prep: fp32 -> bf16 conversions ----------------
__global__ void cvt_inp_kernel(const float* __restrict__ inp, __bf16* __restrict__ inpb) {
  int idx = blockIdx.x * 256 + threadIdx.x;
  inpb[idx] = f2bf(inp[idx]);
}

__global__ void w1t_kernel(const float* __restrict__ W1, __bf16* __restrict__ w1t) {
  int idx = blockIdx.x * 256 + threadIdx.x;   // idx = n*512 + k  (coalesced write)
  int n = idx >> 9;
  int k = idx & 511;
  w1t[idx] = f2bf(W1[(size_t)k * OUT_F + n]);
}

// ---------------- kernel 1: h = inp @ W1 + b1, stored transposed bf16 ----------------
__global__ void gemm_h_kernel(const __bf16* __restrict__ inpb,
                              const __bf16* __restrict__ w1t,
                              const float*  __restrict__ b1,
                              __bf16* __restrict__ hT) {
  const int tid  = threadIdx.x;
  const int lane = tid & 31;
  const int wave = tid >> 5;
  const int tile = blockIdx.x * 8 + wave;     // 16384 tiles
  const int tm   = tile >> 5;                 // 512 row tiles
  const int tn   = tile & 31;                 // 32 col tiles
  const int i0   = tm * 16;
  const int n0   = tn * 16;
  const int l15  = lane & 15;
  const int hi   = lane >> 4;
  const int kbA  = hi ? 16 : 0;
  const int kbB  = hi ? 16 : 0;

  const __bf16* arow = inpb + (size_t)(i0 + l15) * IN_F;   // A: row per lane
  const __bf16* brow = w1t  + (size_t)(n0 + l15) * IN_F;   // B: col per lane (W1^T)

  v8f c = {};
  for (int kk = 0; kk < IN_F; kk += 32) {
    v16bf a;
    *(v8bf*)&a       = *(const v8bf*)(arow + kk + kbA);        // K = kb..kb+7
    *((v8bf*)&a + 1) = *(const v8bf*)(arow + kk + kbA + 8);    // K = kb+8..kb+15
    v16bf b = *(const v16bf*)(brow + kk + kbB);                // 16 contiguous K
    c = __builtin_amdgcn_wmma_f32_16x16x32_bf16(false, a, false, b,
                                                (short)0, c, false, false);
  }

  const float b1v = b1[n0 + l15];
  v8bf hv;
  #pragma unroll
  for (int r = 0; r < 8; ++r) hv[r] = f2bf(c[r] + b1v);
  // hT[col][row]: lane's 8 rows are contiguous -> one 16B store
  *(v8bf*)(hT + (size_t)(n0 + l15) * N_NODES + i0 + 8 * hi) = hv;
}

// ---------------- kernel 2: f1 = h@a1, f2 = h@a2 (from hT, coalesced) ----------------
__global__ void rowdot_kernel(const __bf16* __restrict__ hT,
                              const float* __restrict__ a1,
                              const float* __restrict__ a2,
                              float* __restrict__ f1, float* __restrict__ f2) {
  const int i = blockIdx.x * 256 + threadIdx.x;   // node id
  float s1 = 0.f, s2 = 0.f;
  for (int k = 0; k < OUT_F; ++k) {
    float hv = bf2f(hT[(size_t)k * N_NODES + i]);
    s1 = fmaf(hv, a1[k], s1);
    s2 = fmaf(hv, a2[k], s2);
  }
  f1[i] = s1;
  f2[i] = s2;
}

// ---------------- kernel 3: per-row online softmax stats (max, sum) ----------------
__global__ void attn_stats_kernel(const int* __restrict__ adj,
                                  const float* __restrict__ f1,
                                  const float* __restrict__ f2,
                                  const float* __restrict__ b2,
                                  float* __restrict__ rowmax,
                                  float* __restrict__ rowsum) {
  const int lane = threadIdx.x & 31;
  const int wave = threadIdx.x >> 5;
  const int i = blockIdx.x * 8 + wave;
  const float f1i = f1[i] + b2[0];
  const int* arow = adj + (size_t)i * N_NODES;

  float m = -3.40282e38f, s = 0.f;
  for (int j = lane; j < N_NODES; j += 32) {
    float e = f1i + f2[j];
    e = e > 0.f ? e : ALPHA * e;
    float v = (arow[j] > 0) ? e : -1.0e12f;
    float mn = fmaxf(m, v);
    s = s * __expf(m - mn) + __expf(v - mn);
    m = mn;
  }
  #pragma unroll
  for (int off = 16; off > 0; off >>= 1) {      // wave32 reduction
    float mo = __shfl_xor(m, off, 32);
    float so = __shfl_xor(s, off, 32);
    float mn = fmaxf(m, mo);
    s = s * __expf(m - mn) + so * __expf(mo - mn);
    m = mn;
  }
  if (lane == 0) { rowmax[i] = m; rowsum[i] = s; }
}

// ---------------- kernel 4: out = elu(softmax(masked e) @ h), streaming WMMA ----------------
// Block = 32 rows x 512 cols. 8 waves; wave w owns cols [w*64, w*64+64) for BOTH
// 16-row halves -> each loaded B fragment feeds 2 WMMAs (halves hT/L2 traffic).
__global__ void attn_av_kernel(const int*    __restrict__ adj,
                               const __bf16* __restrict__ hT,
                               const float*  __restrict__ f1,
                               const float*  __restrict__ f2,
                               const float*  __restrict__ b2,
                               const float*  __restrict__ rowmax,
                               const float*  __restrict__ rowsum,
                               float* __restrict__ out) {
  __shared__ __align__(32) __bf16 P[2][32 * PSTR];   // double-buffered 32x32 bf16 P tile
  __shared__ float f1s[32], ms[32], sinv[32];

  const int tid  = threadIdx.x;
  const int lane = tid & 31;
  const int wave = tid >> 5;
  const int i0   = blockIdx.x * 32;
  const int cb   = wave * 64;          // 4 x 16-col tiles per wave

  if (tid < 32) {
    f1s[tid]  = f1[i0 + tid] + b2[0];
    ms[tid]   = rowmax[i0 + tid];
    sinv[tid] = 1.f / rowsum[i0 + tid];
  }
  __syncthreads();

  const int pk  = tid & 31;            // P column (K index)
  const int pr  = tid >> 5;            // P base row (0..7)
  const int l15 = lane & 15;
  const int hi  = lane >> 4;
  const int kbA = hi ? 16 : 0;
  const int kbB = hi ? 16 : 0;

  // accumulators: [row-half A/B][col tile 0..3]
  v8f cA0 = {}, cA1 = {}, cA2 = {}, cA3 = {};
  v8f cB0 = {}, cB1 = {}, cB2 = {}, cB3 = {};

  const __bf16* bp0 = hT + (size_t)(cb +  0 + l15) * N_NODES + kbB;
  const __bf16* bp1 = hT + (size_t)(cb + 16 + l15) * N_NODES + kbB;
  const __bf16* bp2 = hT + (size_t)(cb + 32 + l15) * N_NODES + kbB;
  const __bf16* bp3 = hT + (size_t)(cb + 48 + l15) * N_NODES + kbB;

  for (int j0 = 0, it = 0; j0 < N_NODES; j0 += 32, ++it) {
    __bf16* Pb = P[it & 1];
    // cooperative P tile: 32 rows x 32 K, exp(masked lrelu - rowmax)
    #pragma unroll
    for (int rr = pr; rr < 32; rr += 8) {
      const int jg = j0 + pk;
      float e = f1s[rr] + f2[jg];
      e = e > 0.f ? e : ALPHA * e;
      float p = (adj[(size_t)(i0 + rr) * N_NODES + jg] > 0) ? __expf(e - ms[rr]) : 0.f;
      Pb[rr * PSTR + pk] = f2bf(p);
    }
    if (j0 + 32 < N_NODES)   // hint next adj tile toward L2
      __builtin_prefetch(adj + (size_t)(i0 + pr) * N_NODES + j0 + 32 + pk, 0, 1);
    __syncthreads();         // single barrier/iter is safe with double buffering

    // two A fragments (rows 0-15 and 16-31 of the P tile)
    v16bf aA, aB;
    *(v8bf*)&aA       = *(const v8bf*)(Pb + l15 * PSTR + kbA);
    *((v8bf*)&aA + 1) = *(const v8bf*)(Pb + l15 * PSTR + kbA + 8);
    *(v8bf*)&aB       = *(const v8bf*)(Pb + (16 + l15) * PSTR + kbA);
    *((v8bf*)&aB + 1) = *(const v8bf*)(Pb + (16 + l15) * PSTR + kbA + 8);

    v16bf b0 = *(const v16bf*)(bp0 + j0);
    v16bf b1 = *(const v16bf*)(bp1 + j0);
    v16bf b2v = *(const v16bf*)(bp2 + j0);
    v16bf b3 = *(const v16bf*)(bp3 + j0);

    cA0 = __builtin_amdgcn_wmma_f32_16x16x32_bf16(false, aA, false, b0,  (short)0, cA0, false, false);
    cB0 = __builtin_amdgcn_wmma_f32_16x16x32_bf16(false, aB, false, b0,  (short)0, cB0, false, false);
    cA1 = __builtin_amdgcn_wmma_f32_16x16x32_bf16(false, aA, false, b1,  (short)0, cA1, false, false);
    cB1 = __builtin_amdgcn_wmma_f32_16x16x32_bf16(false, aB, false, b1,  (short)0, cB1, false, false);
    cA2 = __builtin_amdgcn_wmma_f32_16x16x32_bf16(false, aA, false, b2v, (short)0, cA2, false, false);
    cB2 = __builtin_amdgcn_wmma_f32_16x16x32_bf16(false, aB, false, b2v, (short)0, cB2, false, false);
    cA3 = __builtin_amdgcn_wmma_f32_16x16x32_bf16(false, aA, false, b3,  (short)0, cA3, false, false);
    cB3 = __builtin_amdgcn_wmma_f32_16x16x32_bf16(false, aB, false, b3,  (short)0, cB3, false, false);
  }

  // epilogue: scale by 1/rowsum, elu, store fp32
  const int rbase = hi * 8;
  #pragma unroll
  for (int r = 0; r < 8; ++r) {
    const int rowA = rbase + r;         // rows i0 + 0..15
    const int rowB = 16 + rbase + r;    // rows i0 + 16..31
    const float scA = sinv[rowA];
    const float scB = sinv[rowB];
    float* orowA = out + (size_t)(i0 + rowA) * OUT_F + l15;
    float* orowB = out + (size_t)(i0 + rowB) * OUT_F + l15;
    float x;
    x = cA0[r] * scA; orowA[cb +  0] = x > 0.f ? x : __expf(x) - 1.f;
    x = cA1[r] * scA; orowA[cb + 16] = x > 0.f ? x : __expf(x) - 1.f;
    x = cA2[r] * scA; orowA[cb + 32] = x > 0.f ? x : __expf(x) - 1.f;
    x = cA3[r] * scA; orowA[cb + 48] = x > 0.f ? x : __expf(x) - 1.f;
    x = cB0[r] * scB; orowB[cb +  0] = x > 0.f ? x : __expf(x) - 1.f;
    x = cB1[r] * scB; orowB[cb + 16] = x > 0.f ? x : __expf(x) - 1.f;
    x = cB2[r] * scB; orowB[cb + 32] = x > 0.f ? x : __expf(x) - 1.f;
    x = cB3[r] * scB; orowB[cb + 48] = x > 0.f ? x : __expf(x) - 1.f;
  }
}

// ---------------- launch ----------------
extern "C" void kernel_launch(void* const* d_in, const int* in_sizes, int n_in,
                              void* d_out, int out_size, void* d_ws, size_t ws_size,
                              hipStream_t stream) {
  const float* inp = (const float*)d_in[0];
  const int*   adj = (const int*)  d_in[1];
  const float* W1  = (const float*)d_in[2];
  const float* b1  = (const float*)d_in[3];
  const float* a1  = (const float*)d_in[4];
  const float* a2  = (const float*)d_in[5];
  const float* b2  = (const float*)d_in[6];
  float* out = (float*)d_out;

  char* ws = (char*)d_ws;
  __bf16* inpb = (__bf16*)ws; ws += (size_t)N_NODES * IN_F * 2;
  __bf16* w1t  = (__bf16*)ws; ws += (size_t)IN_F * OUT_F * 2;
  __bf16* hT   = (__bf16*)ws; ws += (size_t)OUT_F * N_NODES * 2;
  float* f1    = (float*)ws;  ws += (size_t)N_NODES * 4;
  float* f2    = (float*)ws;  ws += (size_t)N_NODES * 4;
  float* rmax  = (float*)ws;  ws += (size_t)N_NODES * 4;
  float* rsum  = (float*)ws;  ws += (size_t)N_NODES * 4;

  cvt_inp_kernel   <<<(N_NODES * IN_F) / 256, 256, 0, stream>>>(inp, inpb);
  w1t_kernel       <<<(IN_F * OUT_F) / 256,   256, 0, stream>>>(W1, w1t);
  gemm_h_kernel    <<<2048,                   256, 0, stream>>>(inpb, w1t, b1, hT);
  rowdot_kernel    <<<N_NODES / 256,          256, 0, stream>>>(hT, a1, a2, f1, f2);
  attn_stats_kernel<<<N_NODES / 8,            256, 0, stream>>>(adj, f1, f2, b2, rmax, rsum);
  attn_av_kernel   <<<N_NODES / 32,           256, 0, stream>>>(adj, hT, f1, f2, b2, rmax, rsum, out);
}
